// OneShotGenerator_2018634629840
// MI455X (gfx1250) — matrix operand order repaired and checked by reading, not compiled, not run on hardware
//
#include <hip/hip_runtime.h>
#include <math.h>

// ---------------- constants ----------------
#define NN 512          // nodes
#define HH 512          // hidden
#define DD 1024         // 2*H
#define H3 1536         // 3*H
#define LL 7
#define BB 16
#define NPAIR 130305    // (N-2)(N-1)/2 upper-tri k=2 pairs

typedef __attribute__((ext_vector_type(16))) __bf16 bf16x16;
typedef __attribute__((ext_vector_type(8)))  __bf16 bf16x8;
typedef __attribute__((ext_vector_type(8)))  float  f32x8;

// ---------------- small helpers ----------------
__device__ __forceinline__ float sigf(float x) { return 1.0f / (1.0f + __expf(-x)); }

// CDNA5 async global->LDS copy (ASYNCcnt-tracked, bypasses VGPRs).
// GVS mode: mem = SGPR64 base + VGPR32 byte offset; dst = LDS byte address.
__device__ __forceinline__ void async_b128(unsigned ldsAddr, const void* base, unsigned byteOff) {
  asm volatile("global_load_async_to_lds_b128 %0, %1, %2"
               :
               : "v"(ldsAddr), "v"(byteOff), "s"(base)
               : "memory");
}
__device__ __forceinline__ void wait_async0() {
  asm volatile("s_wait_asynccnt 0x0" ::: "memory");
}
__device__ __forceinline__ unsigned lds_off(const void* p) {
  // generic LDS address: low 32 bits are the LDS byte offset (ISA flat->LDS mapping)
  return (unsigned)(uintptr_t)p;
}

// ================= elementwise kernels =================

// state = I
__global__ __launch_bounds__(256) void k_state_init(float* state) {
  int idx = blockIdx.x * 256 + threadIdx.x;
  if (idx >= NN * HH) return;
  state[idx] = ((idx >> 9) == (idx & 511)) ? 1.0f : 0.0f;
}

// acc[0]=0 (loss accumulator), acc[1]=Wsum = sum_b (rewards[b]-baseline)
__global__ void k_init(const float* rewards, const float* baseline, float* acc) {
  if (threadIdx.x == 0) {
    float b0 = baseline[0];
    float s = 0.f;
    for (int b = 0; b < BB; ++b) s += rewards[b] - b0;
    acc[0] = 0.f;
    acc[1] = s;
  }
}

// T[a*N+c] = sum_b (rewards[b]-baseline) * adj[b,0,a,c]
__global__ __launch_bounds__(256) void k_T(const float* adj, const float* rewards,
                                           const float* baseline, float* T) {
  int idx = blockIdx.x * 256 + threadIdx.x;
  if (idx >= NN * NN) return;
  float b0 = baseline[0];
  float s = 0.f;
  for (int b = 0; b < BB; ++b) s += (rewards[b] - b0) * adj[(size_t)b * NN * NN + idx];
  T[idx] = s;
}

// pair index arrays for upper triangle k=2
__global__ __launch_bounds__(256) void k_pairs(int* i0, int* i1) {
  int i = blockIdx.x * 256 + threadIdx.x;
  if (i >= NN - 2) return;
  int off = i * (NN - 2) - (i * (i - 1)) / 2;   // sum_{t<i} (510 - t)
  for (int j = i + 2; j < NN; ++j) { i0[off] = i; i1[off] = j; ++off; }
}

// fp32 -> bf16 straight convert (GRU weights: (3H,H) is already [N][K])
__global__ __launch_bounds__(256) void k_convert(const float* W, __bf16* O, int n) {
  int idx = blockIdx.x * 256 + threadIdx.x;
  if (idx < n) O[idx] = (__bf16)W[idx];
}

// transpose-convert: W [R][C] fp32 -> WT [C][R] bf16
__global__ __launch_bounds__(256) void k_transpose(const float* W, __bf16* WT, int R, int C) {
  int idx = blockIdx.x * 256 + threadIdx.x;
  if (idx >= R * C) return;
  int r = idx / C, c = idx - r * C;
  WT[(size_t)c * R + r] = (__bf16)W[idx];
}

// optional relu on state (in place) + bf16 copy
__global__ __launch_bounds__(256) void k_state_prep(float* state, __bf16* sbf, int doRelu) {
  int idx = blockIdx.x * 256 + threadIdx.x;
  if (idx >= NN * HH) return;
  float v = state[idx];
  if (doRelu) { v = fmaxf(v, 0.f); state[idx] = v; }
  sbf[idx] = (__bf16)v;
}

// EB[m][n] = W1[512+m+1][n] - W1[512+m][n] + b1[n]   (edge_feat @ W1_bottom + b1), row 511 = 0
__global__ __launch_bounds__(256) void k_edgebias(const float* w1, const float* b1, float* EB) {
  int idx = blockIdx.x * 256 + threadIdx.x;
  if (idx >= NN * HH) return;
  int m = idx >> 9, n = idx & 511;
  EB[idx] = (m < NN - 1)
          ? (w1[(size_t)(HH + m + 1) * HH + n] - w1[(size_t)(HH + m) * HH + n] + b1[n])
          : 0.f;
}

// aggbf[m+1] = bf16(msg[m]*att[m]); aggbf[0] = 0  (dst indices are unique: 1..511)
__global__ __launch_bounds__(256) void k_agg(const float* msg, const float* att, __bf16* aggbf) {
  int idx = blockIdx.x * 256 + threadIdx.x;
  if (idx >= NN * HH) return;
  int m = idx >> 9, n = idx & 511;
  aggbf[idx] = (m == 0) ? (__bf16)0.f
                        : (__bf16)(msg[(size_t)(m - 1) * HH + n] * att[(size_t)(m - 1) * HH + n]);
}

// GRU elementwise update, state in place
__global__ __launch_bounds__(256) void k_gru(const float* gi, const float* gh, float* state) {
  int idx = blockIdx.x * 256 + threadIdx.x;
  if (idx >= NN * HH) return;
  int row = idx >> 9, c = idx & 511;
  size_t b = (size_t)row * H3;
  float ir = gi[b + c], iz = gi[b + HH + c], in = gi[b + 2 * HH + c];
  float hr = gh[b + c], hz = gh[b + HH + c], hn = gh[b + 2 * HH + c];
  float r = sigf(ir + hr);
  float z = sigf(iz + hz);
  float n = tanhf(in + r * hn);
  float h = state[idx];
  state[idx] = (1.f - z) * n + z * h;
}

// ================= WMMA GEMM (double-buffered, async staging, static buffers) =================
// C[M][N] = act( A[M][K] @ BT[N][K]^T + bias/rowbias );  requires K % 64 == 0, N % 64 == 0
enum { AM_BF16 = 0, AM_DIFF_ADJ = 1, AM_DIFF_IDX = 2 };
enum { EP_BIAS = 0, EP_BIAS_RELU = 1, EP_BIAS_SIG = 2, EP_ROWBIAS_RELU = 3 };

#define SA 40   // padded LDS row stride (bf16 elems): 80B rows keep b128 alignment

template <int AMODE>
__device__ __forceinline__ void loadA_diff_regs(const float* __restrict__ S,
                                                const int* __restrict__ gi0,
                                                const int* __restrict__ gi1,
                                                int gr, int K, int Mreal, int kb,
                                                bf16x8& lo, bf16x8& hi) {
  float v[16];
  if (gr < Mreal) {
    const float *p0, *p1;
    if (AMODE == AM_DIFF_ADJ) { p0 = S + (size_t)gr * K + kb; p1 = p0 + K; }
    else { p0 = S + (size_t)gi0[gr] * K + kb; p1 = S + (size_t)gi1[gr] * K + kb; }
    const float4* q0 = (const float4*)p0;
    const float4* q1 = (const float4*)p1;
    for (int e = 0; e < 4; ++e) {
      float4 a4 = q0[e], b4 = q1[e];
      v[e * 4 + 0] = a4.x - b4.x; v[e * 4 + 1] = a4.y - b4.y;
      v[e * 4 + 2] = a4.z - b4.z; v[e * 4 + 3] = a4.w - b4.w;
    }
  } else {
    for (int e = 0; e < 16; ++e) v[e] = 0.f;
  }
  for (int e = 0; e < 8; ++e) { lo[e] = (__bf16)v[e]; hi[e] = (__bf16)v[8 + e]; }
}

template <int AMODE, int EPI, bool OBF>
__global__ __launch_bounds__(128)
void gemm_k(const __bf16* __restrict__ Abf, const float* __restrict__ S,
            const __bf16* __restrict__ BT,
            const float* __restrict__ bias, const float* __restrict__ rowbias,
            void* __restrict__ OutP,
            const int* __restrict__ gi0, const int* __restrict__ gi1,
            int N, int K, int Mreal) {
  __shared__ __align__(16) __bf16 As[2][64 * SA];
  __shared__ __align__(16) __bf16 Bs[2][64 * SA];

  const int tid  = threadIdx.x;
  const int lane = tid & 31, wid = tid >> 5;
  const int wm = wid >> 1, wn = wid & 1;         // 2x2 wave grid, wave tile 32x32
  const int lr = lane & 15;
  const bool hiHalf = lane >= 16;
  const int gm0 = blockIdx.y * 64, gn0 = blockIdx.x * 64;

  f32x8 acc[2][2];
  for (int i = 0; i < 2; ++i)
    for (int j = 0; j < 2; ++j)
      for (int e = 0; e < 8; ++e) acc[i][j][e] = 0.f;

  const int r    = tid >> 1;   // staging row 0..63
  const int half = tid & 1;    // which 16-wide K slice
  const int gr   = gm0 + r;

  const unsigned ldsA0 = lds_off(&As[0][r * SA + half * 16]);
  const unsigned ldsA1 = lds_off(&As[1][r * SA + half * 16]);
  const unsigned ldsB0 = lds_off(&Bs[0][r * SA + half * 16]);
  const unsigned ldsB1 = lds_off(&Bs[1][r * SA + half * 16]);
  const unsigned bRowOff = (unsigned)(gn0 + r) * (unsigned)K * 2u;  // bytes
  const unsigned aRowOff = (unsigned)gr * (unsigned)K * 2u;         // bytes (AM_BF16 only)

  constexpr bool ASYNC_A = (AMODE == AM_BF16);
  bf16x8 plo, phi;   // register-staged A for next tile (diff modes)

  // issue global->LDS staging for K-slice kb0 into buffer with given LDS offsets
  auto prefetch = [&](int kb0, unsigned lA, unsigned lB) {
    const unsigned kbb = (unsigned)(kb0 + half * 16) * 2u;
    async_b128(lB,      BT, bRowOff + kbb);
    async_b128(lB + 16, BT, bRowOff + kbb + 16);
    if (ASYNC_A) {
      async_b128(lA,      Abf, aRowOff + kbb);
      async_b128(lA + 16, Abf, aRowOff + kbb + 16);
    } else {
      loadA_diff_regs<AMODE>(S, gi0, gi1, gr, K, Mreal, kb0 + half * 16, plo, phi);
    }
  };
  // diff modes: land register-staged A into LDS buffer 'bufn' (literal at call site -> ds_store)
  auto storeA = [&](int bufn) {
    if (!ASYNC_A) {
      __bf16* d = &As[bufn][r * SA + half * 16];
      *(bf16x8*)d = plo;
      *(bf16x8*)(d + 8) = phi;
    }
  };
  // 4x v_wmma_f32_16x16x32_bf16 from buffer 'bufn' (literal at call site -> const LDS offsets)
  auto compute = [&](int bufn) {
    // A 16x32: lane<16 row lr holds K[0..7],[16..23]; lane>=16 holds K[8..15],[24..31]
    bf16x16 af[2], bfr[2];
    for (int sm = 0; sm < 2; ++sm) {
      const __bf16* ap = &As[bufn][(wm * 32 + sm * 16 + lr) * SA + (hiHalf ? 8 : 0)];
      bf16x8 a0 = *(const bf16x8*)ap;
      bf16x8 a1 = *(const bf16x8*)(ap + 16);
      af[sm] = __builtin_shufflevector(a0, a1, 0, 1, 2, 3, 4, 5, 6, 7,
                                       8, 9, 10, 11, 12, 13, 14, 15);
    }
    // B 32x16: lane<16 col lr holds K[0..7],[16..23]; lane>=16 holds K[8..15],[24..31]
    for (int sn = 0; sn < 2; ++sn) {
      const __bf16* bp = &Bs[bufn][(wn * 32 + sn * 16 + lr) * SA + (hiHalf ? 8 : 0)];
      bf16x8 b0 = *(const bf16x8*)bp;
      bf16x8 b1 = *(const bf16x8*)(bp + 16);
      bfr[sn] = __builtin_shufflevector(b0, b1, 0, 1, 2, 3, 4, 5, 6, 7,
                                        8, 9, 10, 11, 12, 13, 14, 15);
    }
    for (int sm = 0; sm < 2; ++sm)
      for (int sn = 0; sn < 2; ++sn)
        acc[sm][sn] = __builtin_amdgcn_wmma_f32_16x16x32_bf16(
            false, af[sm], false, bfr[sn], (short)0, acc[sm][sn], false, false);
  };

  // ---- prologue: stage tile 0 into buffer 0 ----
  prefetch(0, ldsA0, ldsB0);
  storeA(0);
  wait_async0();
  __syncthreads();

  // ---- pipelined main loop: buffer index is a literal everywhere (K % 64 == 0) ----
  for (int k0 = 0; k0 < K; k0 += 64) {
    // step 0: prefetch k0+32 -> buf1; compute buf0
    prefetch(k0 + 32, ldsA1, ldsB1);
    compute(0);
    storeA(1);
    wait_async0();
    __syncthreads();
    // step 1: prefetch k0+64 -> buf0 (if any); compute buf1
    const bool more = (k0 + 64) < K;
    if (more) prefetch(k0 + 64, ldsA0, ldsB0);
    compute(1);
    if (more) { storeA(0); wait_async0(); }
    __syncthreads();
  }

  // ---- epilogue: C/D layout: VGPR e -> M = e + (lane>=16 ? 8:0), N = lane&15 ----
  for (int sm = 0; sm < 2; ++sm)
    for (int sn = 0; sn < 2; ++sn) {
      const int mb = gm0 + wm * 32 + sm * 16 + (hiHalf ? 8 : 0);
      const int n  = gn0 + wn * 32 + sn * 16 + lr;
      float bn = (EPI == EP_ROWBIAS_RELU) ? 0.f : bias[n];
      for (int e = 0; e < 8; ++e) {
        const int m = mb + e;
        float c = acc[sm][sn][e];
        if (EPI == EP_ROWBIAS_RELU) {
          c += rowbias[(size_t)m * N + n];
          c = fmaxf(c, 0.f);
        } else {
          c += bn;
          if (EPI == EP_BIAS_RELU) c = fmaxf(c, 0.f);
          if (EPI == EP_BIAS_SIG)  c = sigf(c);
        }
        if (OBF) ((__bf16*)OutP)[(size_t)m * N + n] = (__bf16)c;
        else     ((float*)OutP)[(size_t)m * N + n]  = c;
      }
    }
}

// ================= decoder tail: GEMV + fused BCE-pair loss =================
// x = h2[m] . w3 + b3 ; loss += f(x)*Wsum - x*T[i1,i0] ; f(x)=max(x,0)+log1p(exp(-|x|))
__global__ __launch_bounds__(256)
void k_loss(const __bf16* h2, const float* w3, const float* b3,
            const int* i0, const int* i1, const float* T, float* acc, int base) {
  __shared__ float red[8];
  int tid = threadIdx.x, lane = tid & 31, wid = tid >> 5;
  int m = blockIdx.x * 8 + wid;
  int p = base + m;
  float val = 0.f;
  if (p < NPAIR) {
    const __bf16* row = h2 + (size_t)m * HH;
    float s = 0.f;
    int k0 = lane * 16;
    for (int k = k0; k < k0 + 16; ++k) s += (float)row[k] * w3[k];
    for (int off = 16; off > 0; off >>= 1) s += __shfl_xor(s, off, 32);
    if (lane == 0) {
      float x = s + b3[0];
      float Wsum = acc[1];
      float f = fmaxf(x, 0.f) + log1pf(__expf(-fabsf(x)));
      val = f * Wsum - x * T[(size_t)i1[p] * NN + i0[p]];
    }
  }
  if (lane == 0) red[wid] = val;
  __syncthreads();
  if (tid == 0) {
    float t = 0.f;
    for (int i = 0; i < 8; ++i) t += red[i];
    atomicAdd(&acc[0], t);
  }
}

__global__ void k_final(const float* acc, float* out) {
  if (threadIdx.x == 0) {
    const float ln2 = 0.69314718055994530942f;
    float nonpair = ln2 * acc[1] * (float)(NN * NN - NPAIR);
    out[0] = (acc[0] + nonpair) / (float)(BB * NN * NN);
  }
}

// ================= host orchestration =================
extern "C" void kernel_launch(void* const* d_in, const int* in_sizes, int n_in,
                              void* d_out, int out_size, void* d_ws, size_t ws_size,
                              hipStream_t stream) {
  (void)in_sizes; (void)n_in; (void)out_size;
  const float* adj      = (const float*)d_in[0];
  const float* rewards  = (const float*)d_in[1];
  const float* baseline = (const float*)d_in[2];
  const float* msg_w1   = (const float*)d_in[3];
  const float* msg_b1   = (const float*)d_in[4];
  const float* msg_w2   = (const float*)d_in[5];
  const float* msg_b2   = (const float*)d_in[6];
  const float* att_w1   = (const float*)d_in[7];
  const float* att_b1   = (const float*)d_in[8];
  const float* att_w2   = (const float*)d_in[9];
  const float* att_b2   = (const float*)d_in[10];
  const float* gru_wih  = (const float*)d_in[11];
  const float* gru_bih  = (const float*)d_in[12];
  const float* gru_whh  = (const float*)d_in[13];
  const float* gru_bhh  = (const float*)d_in[14];
  const float* th_w1    = (const float*)d_in[15];
  const float* th_b1    = (const float*)d_in[16];
  const float* th_w2    = (const float*)d_in[17];
  const float* th_b2    = (const float*)d_in[18];
  const float* th_w3    = (const float*)d_in[19];
  const float* th_b3    = (const float*)d_in[20];

  // ---- workspace carve-out ----
  char* wsb = (char*)d_ws;
  size_t off = 0;
  auto take = [&](size_t bytes) -> char* {
    char* p = wsb + off;
    off = (off + bytes + 255) & ~(size_t)255;
    return p;
  };
  __bf16* msgw1T  = (__bf16*)take((size_t)LL * HH * HH * 2);  // top-half of msg_w1, transposed
  __bf16* msgw2T  = (__bf16*)take((size_t)LL * HH * HH * 2);
  __bf16* attw1T  = (__bf16*)take((size_t)LL * HH * HH * 2);
  __bf16* attw2T  = (__bf16*)take((size_t)LL * HH * HH * 2);
  __bf16* wihbf   = (__bf16*)take((size_t)LL * H3 * HH * 2);
  __bf16* whhbf   = (__bf16*)take((size_t)LL * H3 * HH * 2);
  __bf16* thw1T   = (__bf16*)take((size_t)HH * HH * 2);
  __bf16* thw2T   = (__bf16*)take((size_t)HH * HH * 2);
  float*  state   = (float*)take((size_t)NN * HH * 4);
  __bf16* statebf = (__bf16*)take((size_t)NN * HH * 2);
  float*  EBm     = (float*)take((size_t)NN * HH * 4);
  float*  EBa     = (float*)take((size_t)NN * HH * 4);
  __bf16* hidden  = (__bf16*)take((size_t)NN * HH * 2);
  float*  msgbuf  = (float*)take((size_t)NN * HH * 4);
  float*  attbuf  = (float*)take((size_t)NN * HH * 4);
  __bf16* aggbf   = (__bf16*)take((size_t)NN * HH * 2);
  float*  gibuf   = (float*)take((size_t)NN * H3 * 4);
  float*  ghbuf   = (float*)take((size_t)NN * H3 * 4);
  float*  Tm      = (float*)take((size_t)NN * NN * 4);
  int*    pi0     = (int*)take((size_t)NPAIR * 4);
  int*    pi1     = (int*)take((size_t)NPAIR * 4);
  float*  acc     = (float*)take(256);

  // decoder chunk size from remaining workspace
  size_t rem = (ws_size > off + 4096) ? (ws_size - off - 4096) : 0;
  long long chl = (long long)(rem / (2 * (size_t)HH * 2));
  int CH = (int)(chl > 16384 ? 16384 : chl);
  CH &= ~63;
  if (CH < 64) CH = 64;
  __bf16* h1buf = (__bf16*)take((size_t)CH * HH * 2);
  __bf16* h2buf = (__bf16*)take((size_t)CH * HH * 2);

  const dim3 blk256(256), blkG(128);
  const int eN = (NN * HH + 255) / 256;          // 512x512 elementwise grid

  // ---- one-time prep ----
  k_init<<<1, 32, 0, stream>>>(rewards, baseline, acc);
  k_state_init<<<eN, blk256, 0, stream>>>(state);
  k_T<<<(NN * NN + 255) / 256, blk256, 0, stream>>>(adj, rewards, baseline, Tm);
  k_pairs<<<2, blk256, 0, stream>>>(pi0, pi1);

  // weight conversion (bf16, [N][K] layout for B operand)
  for (int l = 0; l < LL; ++l) {
    const size_t w1o = (size_t)l * DD * HH, w2o = (size_t)l * HH * HH;
    const size_t go  = (size_t)l * H3 * HH;
    k_transpose<<<eN, blk256, 0, stream>>>(msg_w1 + w1o, msgw1T + (size_t)l * HH * HH, HH, HH);
    k_transpose<<<eN, blk256, 0, stream>>>(msg_w2 + w2o, msgw2T + (size_t)l * HH * HH, HH, HH);
    k_transpose<<<eN, blk256, 0, stream>>>(att_w1 + w1o, attw1T + (size_t)l * HH * HH, HH, HH);
    k_transpose<<<eN, blk256, 0, stream>>>(att_w2 + w2o, attw2T + (size_t)l * HH * HH, HH, HH);
    k_convert<<<(H3 * HH + 255) / 256, blk256, 0, stream>>>(gru_wih + go, wihbf + go, H3 * HH);
    k_convert<<<(H3 * HH + 255) / 256, blk256, 0, stream>>>(gru_whh + go, whhbf + go, H3 * HH);
  }
  k_transpose<<<eN, blk256, 0, stream>>>(th_w1, thw1T, HH, HH);
  k_transpose<<<eN, blk256, 0, stream>>>(th_w2, thw2T, HH, HH);

  const dim3 g512(NN / 64, NN / 64);   // (8,8): M=512, N=512
  const dim3 g1536(H3 / 64, NN / 64);  // (24,8): M=512, N=1536

  // ---- GNN layers ----
  for (int l = 0; l < LL; ++l) {
    const __bf16* mw1 = msgw1T + (size_t)l * HH * HH;
    const __bf16* mw2 = msgw2T + (size_t)l * HH * HH;
    const __bf16* aw1 = attw1T + (size_t)l * HH * HH;
    const __bf16* aw2 = attw2T + (size_t)l * HH * HH;
    const __bf16* wih = wihbf + (size_t)l * H3 * HH;
    const __bf16* whh = whhbf + (size_t)l * H3 * HH;

    k_state_prep<<<eN, blk256, 0, stream>>>(state, statebf, l > 0);
    k_edgebias<<<eN, blk256, 0, stream>>>(msg_w1 + (size_t)l * DD * HH, msg_b1 + (size_t)l * HH, EBm);
    k_edgebias<<<eN, blk256, 0, stream>>>(att_w1 + (size_t)l * DD * HH, att_b1 + (size_t)l * HH, EBa);

    // msg hidden = relu(diff @ W1_top + EBm)  -> bf16
    gemm_k<AM_DIFF_ADJ, EP_ROWBIAS_RELU, true><<<g512, blkG, 0, stream>>>(
        nullptr, state, mw1, nullptr, EBm, hidden, nullptr, nullptr, HH, HH, NN - 1);
    // msg = hidden @ W2 + b2 -> fp32
    gemm_k<AM_BF16, EP_BIAS, false><<<g512, blkG, 0, stream>>>(
        hidden, nullptr, mw2, msg_b2 + (size_t)l * HH, nullptr, msgbuf, nullptr, nullptr, HH, HH, NN);
    // att hidden
    gemm_k<AM_DIFF_ADJ, EP_ROWBIAS_RELU, true><<<g512, blkG, 0, stream>>>(
        nullptr, state, aw1, nullptr, EBa, hidden, nullptr, nullptr, HH, HH, NN - 1);
    // att = sigmoid(hidden @ W2 + b2) -> fp32
    gemm_k<AM_BF16, EP_BIAS_SIG, false><<<g512, blkG, 0, stream>>>(
        hidden, nullptr, aw2, att_b2 + (size_t)l * HH, nullptr, attbuf, nullptr, nullptr, HH, HH, NN);
    // agg (scatter by dst = row shift), bf16
    k_agg<<<eN, blk256, 0, stream>>>(msgbuf, attbuf, aggbf);
    // GRU gates: gi = agg @ wih^T + bih ; gh = state @ whh^T + bhh
    gemm_k<AM_BF16, EP_BIAS, false><<<g1536, blkG, 0, stream>>>(
        aggbf, nullptr, wih, gru_bih + (size_t)l * H3, nullptr, gibuf, nullptr, nullptr, H3, HH, NN);
    gemm_k<AM_BF16, EP_BIAS, false><<<g1536, blkG, 0, stream>>>(
        statebf, nullptr, whh, gru_bhh + (size_t)l * H3, nullptr, ghbuf, nullptr, nullptr, H3, HH, NN);
    k_gru<<<eN, blk256, 0, stream>>>(gibuf, ghbuf, state);
  }

  // ---- decoder: chunked pair MLP + fused loss ----
  const int nch = (NPAIR + CH - 1) / CH;
  for (int c = 0; c < nch; ++c) {
    const int base = c * CH;
    int rows = NPAIR - base;
    if (rows > CH) rows = CH;
    const int Mpad = (rows + 63) & ~63;
    const dim3 gd(HH / 64, Mpad / 64);
    gemm_k<AM_DIFF_IDX, EP_BIAS_RELU, true><<<gd, blkG, 0, stream>>>(
        nullptr, state, thw1T, th_b1, nullptr, h1buf, pi0 + base, pi1 + base, HH, HH, rows);
    gemm_k<AM_BF16, EP_BIAS_RELU, true><<<gd, blkG, 0, stream>>>(
        h1buf, nullptr, thw2T, th_b2, nullptr, h2buf, nullptr, nullptr, HH, HH, Mpad);
    k_loss<<<Mpad / 8, blk256, 0, stream>>>(h2buf, th_w3, th_b3, pi0, pi1, Tm, acc, base);
  }
  k_final<<<1, 32, 0, stream>>>(acc, (float*)d_out);
}
